// WordEncoder_37950331027936
// MI455X (gfx1250) — compile-verified, matrix operand order checked by reading, not run on hardware
//
#include <hip/hip_runtime.h>
#include <hip/hip_bf16.h>

// ---------------------------------------------------------------------------
// CDNA5 (gfx1250) WordEncoder: all GEMMs via v_wmma_f32_16x16x32_bf16.
// wave32: one wave (32 threads) per 32x32 output tile (2x2 WMMA register
// blocking -> 4 wmma per 8 b128 loads). EXEC all ones around every WMMA.
// ---------------------------------------------------------------------------

typedef __bf16 bf16_t;
typedef __attribute__((ext_vector_type(16))) __bf16 v16bf;
typedef __attribute__((ext_vector_type(8)))  float  v8f;

#define B_   16
#define S_   512
#define H_   768
#define W_   128
#define NH_  4
#define DH_  192
#define FF_  2048
#define EPS_ 1e-5f

union FragU { v16bf v; unsigned u[8]; };

// A-fragment (16x32 bf16, M on lanes 0..15) per CDNA5 ISA:
//   lane<16 : VGPR v holds K = {2v,2v+1} for v<4, {8+2v,9+2v} for v>=4
//   lane>=16: same pattern + 8
// B mirrors with lane = N, loaded from an (N x K) row-major buffer.
__device__ __forceinline__ v16bf load_frag_row(const bf16_t* rowbase, int k0, int lane) {
  FragU f;
  int hi = (lane >> 4) * 8;
#pragma unroll
  for (int v = 0; v < 8; ++v) {
    int k = k0 + hi + (v < 4 ? 2 * v : 8 + 2 * v);
    f.u[v] = *(const unsigned*)(rowbase + k);   // pair (k, k+1), low half = k
  }
  return f.v;
}

__device__ __forceinline__ v8f wmma_bf16(v16bf a, v16bf b, v8f c) {
  // (neg_a, A, neg_b, B, c_mod, C, reuse_a, reuse_b)
  return __builtin_amdgcn_wmma_f32_16x16x32_bf16(false, a, false, b,
                                                 (short)0, c, false, false);
}

struct Acc4 { v8f a00, a01, a10, a11; };

// 32x32 tile mainloop: A0/A1 = rows m..m+15 / m+16..m+31 (stride K),
// B0/B1 = cols n..n+15 / n+16..n+31 from (N,K) buffer.
__device__ __forceinline__ Acc4 wmma_kloop_2x2(const bf16_t* A0, const bf16_t* A1,
                                               const bf16_t* B0, const bf16_t* B1,
                                               int K, int lane) {
  v8f z = {0.f, 0.f, 0.f, 0.f, 0.f, 0.f, 0.f, 0.f};
  Acc4 acc = {z, z, z, z};
  for (int k0 = 0; k0 < K; k0 += 32) {
    v16bf a0 = load_frag_row(A0, k0, lane);
    v16bf a1 = load_frag_row(A1, k0, lane);
    v16bf b0 = load_frag_row(B0, k0, lane);
    v16bf b1 = load_frag_row(B1, k0, lane);
    acc.a00 = wmma_bf16(a0, b0, acc.a00);
    acc.a01 = wmma_bf16(a0, b1, acc.a01);
    acc.a10 = wmma_bf16(a1, b0, acc.a10);
    acc.a11 = wmma_bf16(a1, b1, acc.a11);
  }
  return acc;
}

// ---------------------------------------------------------------------------
// 1) weight f32 (K,N) row-major  ->  bf16 (N,K) row-major ("BT")
// ---------------------------------------------------------------------------
__global__ void convert_T_kernel(const float* __restrict__ in, bf16_t* __restrict__ out,
                                 int K, int N) {
  long long i = (long long)blockIdx.x * blockDim.x + threadIdx.x;
  if (i >= (long long)K * N) return;
  int k = (int)(i / N), n = (int)(i % N);
  out[(size_t)n * K + k] = (bf16_t)in[i];
}

// ---------------------------------------------------------------------------
// 2) embedding gather + segment mean (word_ids sorted per row -> binary search)
// ---------------------------------------------------------------------------
__global__ void seg_mean_kernel(const int* __restrict__ ids, const int* __restrict__ mask,
                                const int* __restrict__ wid, const float* __restrict__ emb,
                                float* __restrict__ outF, bf16_t* __restrict__ outB) {
  int bw = blockIdx.x;
  int b = bw / W_, w = bw % W_;
  const int* row = wid + (size_t)b * S_;
  int lo = 0, hi = S_;
  while (lo < hi) { int m = (lo + hi) >> 1; if (row[m] <  w) lo = m + 1; else hi = m; }
  int start = lo;
  lo = start; hi = S_;
  while (lo < hi) { int m = (lo + hi) >> 1; if (row[m] <= w) lo = m + 1; else hi = m; }
  int end = lo;
  int cnt = end - start;
  float inv = (cnt > 0) ? 1.f / (float)cnt : 0.f;
  for (int h = threadIdx.x; h < H_; h += blockDim.x) {
    float s = 0.f;
    for (int t = start; t < end; ++t) {
      int tok = ids[(size_t)b * S_ + t];
      float mk = (float)mask[(size_t)b * S_ + t];
      s += emb[(size_t)tok * H_ + h] * mk;
    }
    float mean = s * inv;
    size_t o = ((size_t)b * W_ + w) * H_ + h;
    outF[o] = mean;
    outB[o] = (bf16_t)mean;
  }
}

// ---------------------------------------------------------------------------
// 3) QKV GEMM: (2048,768) @ (768,2304) + bias; scatter q/k row-major per head,
//    v transposed per head (so attn GEMMs consume them directly as WMMA frags)
// ---------------------------------------------------------------------------
__device__ __forceinline__ void qkv_store(float accv, int m, int n,
                                          const float* __restrict__ bias,
                                          bf16_t* __restrict__ qb,
                                          bf16_t* __restrict__ kb,
                                          bf16_t* __restrict__ vT) {
  int part = n / H_;        // 0=q 1=k 2=v
  int rem  = n % H_;
  int head = rem / DH_;
  int d    = rem % DH_;
  int b = m / W_, w = m % W_;
  bf16_t val = (bf16_t)(accv + bias[n]);
  size_t bh = (size_t)(b * NH_ + head);
  if (part == 0)      qb[(bh * W_  + w) * DH_ + d] = val;
  else if (part == 1) kb[(bh * W_  + w) * DH_ + d] = val;
  else                vT[(bh * DH_ + d) * W_  + w] = val;
}

__global__ __launch_bounds__(32) void gemm_qkv_kernel(
    const bf16_t* __restrict__ A, const bf16_t* __restrict__ BT,
    const float* __restrict__ bias,
    bf16_t* __restrict__ qb, bf16_t* __restrict__ kb, bf16_t* __restrict__ vT) {
  int lane = threadIdx.x;
  const int Mt = (B_ * W_) / 32;
  int tm = blockIdx.x % Mt;
  int tn = blockIdx.x / Mt;
  const bf16_t* A0 = A  + (size_t)(tm * 32 + (lane & 15)) * H_;
  const bf16_t* B0 = BT + (size_t)(tn * 32 + (lane & 15)) * H_;
  Acc4 acc = wmma_kloop_2x2(A0, A0 + (size_t)16 * H_, B0, B0 + (size_t)16 * H_, H_, lane);
  int n0  = tn * 32 + (lane & 15), n1 = n0 + 16;
  int mb0 = tm * 32 + ((lane >> 4) * 8), mb1 = mb0 + 16;
#pragma unroll
  for (int r = 0; r < 8; ++r) {
    qkv_store(acc.a00[r], mb0 + r, n0, bias, qb, kb, vT);
    qkv_store(acc.a01[r], mb0 + r, n1, bias, qb, kb, vT);
    qkv_store(acc.a10[r], mb1 + r, n0, bias, qb, kb, vT);
    qkv_store(acc.a11[r], mb1 + r, n1, bias, qb, kb, vT);
  }
}

// ---------------------------------------------------------------------------
// 4) generic batched GEMM: out = A(M,K) @ BT(N,K)^T * scale + bias + resid,
//    optional relu, optional f32 and/or bf16 outputs. 32x32 tile per wave.
// ---------------------------------------------------------------------------
__device__ __forceinline__ void gemm_store(
    float accv, int m, int n, int batch,
    const float* __restrict__ bias,
    const float* __restrict__ resid, long long sR, int ldR,
    float* __restrict__ outF, long long sOF,
    bf16_t* __restrict__ outB, long long sOB,
    int ldO, int relu, float scale) {
  float val = accv * scale + (bias ? bias[n] : 0.f);
  if (resid) val += resid[(size_t)batch * sR + (size_t)m * ldR + n];
  if (relu)  val = fmaxf(val, 0.f);
  size_t oi = (size_t)m * ldO + n;
  if (outF) outF[(size_t)batch * sOF + oi] = val;
  if (outB) outB[(size_t)batch * sOB + oi] = (bf16_t)val;
}

__global__ __launch_bounds__(32) void gemm_bf16_kernel(
    const bf16_t* __restrict__ A, long long sA,
    const bf16_t* __restrict__ BT, long long sB,
    const float* __restrict__ bias,
    const float* __restrict__ resid, long long sR, int ldR,
    float* __restrict__ outF, long long sOF,
    bf16_t* __restrict__ outB, long long sOB,
    int ldO, int Mt32, int K, int relu, float scale) {
  int lane  = threadIdx.x;
  int batch = blockIdx.y;
  int tm = blockIdx.x % Mt32;
  int tn = blockIdx.x / Mt32;
  const bf16_t* A0 = A  + (size_t)batch * sA + (size_t)(tm * 32 + (lane & 15)) * K;
  const bf16_t* B0 = BT + (size_t)batch * sB + (size_t)(tn * 32 + (lane & 15)) * K;
  Acc4 acc = wmma_kloop_2x2(A0, A0 + (size_t)16 * K, B0, B0 + (size_t)16 * K, K, lane);
  int n0  = tn * 32 + (lane & 15), n1 = n0 + 16;
  int mb0 = tm * 32 + ((lane >> 4) * 8), mb1 = mb0 + 16;
#pragma unroll
  for (int r = 0; r < 8; ++r) {
    gemm_store(acc.a00[r], mb0 + r, n0, batch, bias, resid, sR, ldR, outF, sOF, outB, sOB, ldO, relu, scale);
    gemm_store(acc.a01[r], mb0 + r, n1, batch, bias, resid, sR, ldR, outF, sOF, outB, sOB, ldO, relu, scale);
    gemm_store(acc.a10[r], mb1 + r, n0, batch, bias, resid, sR, ldR, outF, sOF, outB, sOB, ldO, relu, scale);
    gemm_store(acc.a11[r], mb1 + r, n1, batch, bias, resid, sR, ldR, outF, sOF, outB, sOB, ldO, relu, scale);
  }
}

// ---------------------------------------------------------------------------
// 5) softmax over rows of 128 (one block = one row)
// ---------------------------------------------------------------------------
__global__ void softmax_kernel(const float* __restrict__ scores, bf16_t* __restrict__ attn) {
  __shared__ float red[W_];
  int row = blockIdx.x, t = threadIdx.x;
  float v = scores[(size_t)row * W_ + t];
  red[t] = v; __syncthreads();
  for (int s = W_ / 2; s > 0; s >>= 1) {
    if (t < s) red[t] = fmaxf(red[t], red[t + s]);
    __syncthreads();
  }
  float mx = red[0]; __syncthreads();
  float e = __expf(v - mx);
  red[t] = e; __syncthreads();
  for (int s = W_ / 2; s > 0; s >>= 1) {
    if (t < s) red[t] += red[t + s];
    __syncthreads();
  }
  attn[(size_t)row * W_ + t] = (bf16_t)(e / red[0]);
}

// ---------------------------------------------------------------------------
// 6) ctx = attn(W,W) @ v(W,DH) per (b,h); vT(DH,W) is the WMMA B layout.
//    Scatter result into (B*W, H) bf16 for the w_o GEMM. 32x32 tiles.
// ---------------------------------------------------------------------------
__global__ __launch_bounds__(32) void gemm_ctx_kernel(
    const bf16_t* __restrict__ attn, const bf16_t* __restrict__ vT,
    bf16_t* __restrict__ ctx) {
  int lane = threadIdx.x;
  int bh = blockIdx.y;
  const int Mt = W_ / 32;
  int tm = blockIdx.x % Mt;
  int tn = blockIdx.x / Mt;
  const bf16_t* A0 = attn + (size_t)bh * W_ * W_  + (size_t)(tm * 32 + (lane & 15)) * W_;
  const bf16_t* B0 = vT   + (size_t)bh * DH_ * W_ + (size_t)(tn * 32 + (lane & 15)) * W_;
  Acc4 acc = wmma_kloop_2x2(A0, A0 + (size_t)16 * W_, B0, B0 + (size_t)16 * W_, W_, lane);
  int b = bh / NH_, h = bh % NH_;
  int n0  = tn * 32 + (lane & 15), n1 = n0 + 16;
  int mb0 = tm * 32 + ((lane >> 4) * 8), mb1 = mb0 + 16;
#pragma unroll
  for (int r = 0; r < 8; ++r) {
    ctx[((size_t)(b * W_ + mb0 + r)) * H_ + h * DH_ + n0] = (bf16_t)acc.a00[r];
    ctx[((size_t)(b * W_ + mb0 + r)) * H_ + h * DH_ + n1] = (bf16_t)acc.a01[r];
    ctx[((size_t)(b * W_ + mb1 + r)) * H_ + h * DH_ + n0] = (bf16_t)acc.a10[r];
    ctx[((size_t)(b * W_ + mb1 + r)) * H_ + h * DH_ + n1] = (bf16_t)acc.a11[r];
  }
}

// ---------------------------------------------------------------------------
// 7) LayerNorm over H=768 (one block = one row)
// ---------------------------------------------------------------------------
__global__ void layernorm_kernel(const float* __restrict__ in, const float* __restrict__ g,
                                 const float* __restrict__ be, float* __restrict__ outF,
                                 bf16_t* __restrict__ outB) {
  __shared__ float red[256];
  int row = blockIdx.x, t = threadIdx.x;
  const float* x = in + (size_t)row * H_;
  float s = 0.f;
  for (int h = t; h < H_; h += 256) s += x[h];
  red[t] = s; __syncthreads();
  for (int st = 128; st > 0; st >>= 1) { if (t < st) red[t] += red[t + st]; __syncthreads(); }
  float mu = red[0] / (float)H_; __syncthreads();
  float v = 0.f;
  for (int h = t; h < H_; h += 256) { float d = x[h] - mu; v += d * d; }
  red[t] = v; __syncthreads();
  for (int st = 128; st > 0; st >>= 1) { if (t < st) red[t] += red[t + st]; __syncthreads(); }
  float rstd = rsqrtf(red[0] / (float)H_ + EPS_);
  for (int h = t; h < H_; h += 256) {
    float y = (x[h] - mu) * rstd * g[h] + be[h];
    outF[(size_t)row * H_ + h] = y;
    if (outB) outB[(size_t)row * H_ + h] = (bf16_t)y;
  }
}

// ---------------------------------------------------------------------------
// 8) pooled = x2[:,0,:]; logits = pooled @ w_c(768,2) + b_c
//    d_out layout: [logits (16*2)] then [pooled (16*768)]
// ---------------------------------------------------------------------------
__global__ void classifier_kernel(const float* __restrict__ x2, const float* __restrict__ wc,
                                  const float* __restrict__ bc, float* __restrict__ out) {
  __shared__ float r0[256], r1[256];
  int b = blockIdx.x, t = threadIdx.x;
  const float* row = x2 + (size_t)b * W_ * H_;   // word 0
  float s0 = 0.f, s1 = 0.f;
  for (int h = t; h < H_; h += 256) {
    float xv = row[h];
    s0 += xv * wc[h * 2 + 0];
    s1 += xv * wc[h * 2 + 1];
    out[B_ * 2 + (size_t)b * H_ + h] = xv;       // pooled
  }
  r0[t] = s0; r1[t] = s1; __syncthreads();
  for (int st = 128; st > 0; st >>= 1) {
    if (t < st) { r0[t] += r0[t + st]; r1[t] += r1[t + st]; }
    __syncthreads();
  }
  if (t == 0) { out[b * 2 + 0] = r0[0] + bc[0]; out[b * 2 + 1] = r1[0] + bc[1]; }
}

// ---------------------------------------------------------------------------
extern "C" void kernel_launch(void* const* d_in, const int* in_sizes, int n_in,
                              void* d_out, int out_size, void* d_ws, size_t ws_size,
                              hipStream_t stream) {
  const int*   input_ids = (const int*)  d_in[0];
  const int*   attn_mask = (const int*)  d_in[1];
  const int*   word_ids  = (const int*)  d_in[2];
  const float* emb       = (const float*)d_in[3];
  const float* w_qkv     = (const float*)d_in[4];
  const float* b_qkv     = (const float*)d_in[5];
  const float* w_o       = (const float*)d_in[6];
  const float* b_o       = (const float*)d_in[7];
  const float* ln1_g     = (const float*)d_in[8];
  const float* ln1_b     = (const float*)d_in[9];
  const float* w1        = (const float*)d_in[10];
  const float* b1        = (const float*)d_in[11];
  const float* w2        = (const float*)d_in[12];
  const float* b2        = (const float*)d_in[13];
  const float* ln2_g     = (const float*)d_in[14];
  const float* ln2_b     = (const float*)d_in[15];
  const float* w_c       = (const float*)d_in[16];
  const float* b_c       = (const float*)d_in[17];
  float* out = (float*)d_out;

  // bump allocator over d_ws (256B aligned)
  size_t off = 0;
  auto alloc = [&](size_t bytes) -> void* {
    off = (off + 255) & ~(size_t)255;
    void* p = (char*)d_ws + off;
    off += bytes;
    return p;
  };
  const size_t MH   = (size_t)B_ * W_ * H_;
  const size_t BHWD = (size_t)B_ * NH_ * W_ * DH_;
  const size_t SC   = (size_t)B_ * NH_ * W_ * W_;

  float*  padded_f  = (float*) alloc(MH * 4);
  bf16_t* padded_bf = (bf16_t*)alloc(MH * 2);
  bf16_t* wTqkv     = (bf16_t*)alloc((size_t)3 * H_ * H_ * 2);
  bf16_t* wTo       = (bf16_t*)alloc((size_t)H_ * H_ * 2);
  bf16_t* w1T       = (bf16_t*)alloc((size_t)FF_ * H_ * 2);
  bf16_t* w2T       = (bf16_t*)alloc((size_t)H_ * FF_ * 2);
  bf16_t* qb        = (bf16_t*)alloc(BHWD * 2);
  bf16_t* kb        = (bf16_t*)alloc(BHWD * 2);
  bf16_t* vT        = (bf16_t*)alloc(BHWD * 2);
  float*  scores    = (float*) alloc(SC * 4);
  bf16_t* attn_bf   = (bf16_t*)alloc(SC * 2);
  bf16_t* ctx_bf    = (bf16_t*)alloc(MH * 2);
  float*  y1        = (float*) alloc(MH * 4);
  float*  x1_f      = (float*) alloc(MH * 4);
  bf16_t* x1_bf     = (bf16_t*)alloc(MH * 2);
  bf16_t* ff1_bf    = (bf16_t*)alloc((size_t)B_ * W_ * FF_ * 2);
  float*  y2        = (float*) alloc(MH * 4);
  float*  x2_f      = (float*) alloc(MH * 4);
  (void)ws_size; (void)n_in; (void)in_sizes; (void)out_size;

  // weight conversion to bf16 (N,K)
  {
    long long t;
    t = (long long)H_ * (3 * H_);
    convert_T_kernel<<<(unsigned)((t + 255) / 256), 256, 0, stream>>>(w_qkv, wTqkv, H_, 3 * H_);
    t = (long long)H_ * H_;
    convert_T_kernel<<<(unsigned)((t + 255) / 256), 256, 0, stream>>>(w_o, wTo, H_, H_);
    t = (long long)H_ * FF_;
    convert_T_kernel<<<(unsigned)((t + 255) / 256), 256, 0, stream>>>(w1, w1T, H_, FF_);
    t = (long long)FF_ * H_;
    convert_T_kernel<<<(unsigned)((t + 255) / 256), 256, 0, stream>>>(w2, w2T, FF_, H_);
  }

  // embedding gather + segment mean
  seg_mean_kernel<<<B_ * W_, 256, 0, stream>>>(input_ids, attn_mask, word_ids, emb,
                                               padded_f, padded_bf);

  // QKV projection (WMMA, 32x32 tiles)
  {
    int Mt = (B_ * W_) / 32, Nt = (3 * H_) / 32;
    gemm_qkv_kernel<<<Mt * Nt, 32, 0, stream>>>(padded_bf, wTqkv, b_qkv, qb, kb, vT);
  }

  // scores = q @ k^T / sqrt(DH)  (batched WMMA over B*NH)
  {
    dim3 grid((W_ / 32) * (W_ / 32), B_ * NH_);
    gemm_bf16_kernel<<<grid, 32, 0, stream>>>(
        qb, (long long)W_ * DH_, kb, (long long)W_ * DH_,
        nullptr, nullptr, 0, 0,
        scores, (long long)W_ * W_, nullptr, 0,
        W_, W_ / 32, DH_, 0, 0.07216878364870322f /* 1/sqrt(192) */);
  }

  // softmax rows
  softmax_kernel<<<B_ * NH_ * W_, W_, 0, stream>>>(scores, attn_bf);

  // ctx = attn @ v  (batched WMMA), scatter to (B*W, H)
  {
    dim3 grid((W_ / 32) * (DH_ / 32), B_ * NH_);
    gemm_ctx_kernel<<<grid, 32, 0, stream>>>(attn_bf, vT, ctx_bf);
  }

  // y1 = padded + ctx @ w_o + b_o
  {
    dim3 grid(((B_ * W_) / 32) * (H_ / 32), 1);
    gemm_bf16_kernel<<<grid, 32, 0, stream>>>(
        ctx_bf, 0, wTo, 0, b_o,
        padded_f, 0, H_,
        y1, 0, nullptr, 0,
        H_, (B_ * W_) / 32, H_, 0, 1.0f);
  }

  // x1 = LN1(y1)
  layernorm_kernel<<<B_ * W_, 256, 0, stream>>>(y1, ln1_g, ln1_b, x1_f, x1_bf);

  // ff1 = relu(x1 @ w1 + b1)
  {
    dim3 grid(((B_ * W_) / 32) * (FF_ / 32), 1);
    gemm_bf16_kernel<<<grid, 32, 0, stream>>>(
        x1_bf, 0, w1T, 0, b1,
        nullptr, 0, 0,
        nullptr, 0, ff1_bf, 0,
        FF_, (B_ * W_) / 32, H_, 1, 1.0f);
  }

  // y2 = x1 + ff1 @ w2 + b2
  {
    dim3 grid(((B_ * W_) / 32) * (H_ / 32), 1);
    gemm_bf16_kernel<<<grid, 32, 0, stream>>>(
        ff1_bf, 0, w2T, 0, b2,
        x1_f, 0, H_,
        y2, 0, nullptr, 0,
        H_, (B_ * W_) / 32, FF_, 0, 1.0f);
  }

  // x2 = LN2(y2)
  layernorm_kernel<<<B_ * W_, 256, 0, stream>>>(y2, ln2_g, ln2_b, x2_f, nullptr);

  // logits + pooled
  classifier_kernel<<<B_, 256, 0, stream>>>(x2_f, w_c, b_c, out);
}
